// gcn_69458211111595
// MI455X (gfx1250) — matrix operand order repaired and checked by reading, not compiled, not run on hardware
//
#include <hip/hip_runtime.h>

// ---------------------------------------------------------------------------
// Fused GCN block for MI455X (gfx1250, wave32, WMMA bf16 16x16x32).
// One workgroup per (b, m) task; 16 waves of 32 lanes.
// Round 2: base-matrix B-fragments held register-resident across the
// order-1 and order-2 nconvs -> A read from HBM exactly once per task.
// ---------------------------------------------------------------------------

typedef __attribute__((ext_vector_type(16))) __bf16 v16bf;
typedef __attribute__((ext_vector_type(8)))  float  v8f;
typedef __attribute__((ext_vector_type(4)))  unsigned int v4u;

union FragU { v16bf v; __bf16 h[16]; v4u q[2]; };

#define HN_PITCH 264   // halves; 528B rows -> b128 frag loads spread across banks
#define HT_PITCH 232   // halves; 464B rows

static __device__ __forceinline__ v8f wmma_bf16(v16bf a, v16bf b, v8f c) {
  // D = A x B + C, f32 accumulate. 8 args: neg_a, A, neg_b, B, c_mod, C, reuse_a, reuse_b
  return __builtin_amdgcn_wmma_f32_16x16x32_bf16(false, a, false, b, (short)0, c,
                                                 false, false);
}

// A-operand (16xK tile of an MxK matrix) from bf16 LDS, row-major [row][k].
// Lane holds row (lane&15); 16 contiguous K values starting at k0+kh*16 (same order as B).
static __device__ __forceinline__ v16bf afrag_lds(const __bf16* S, int pitch,
                                                  int row_base, int k0, int lane) {
  const int r = lane & 15, kh = lane >> 4;
  const __bf16* p = S + (row_base + r) * pitch + k0 + kh * 16;
  FragU f;
  f.q[0] = *(const v4u*)(p);        // K = k0+kh*16 .. +7
  f.q[1] = *(const v4u*)(p + 8);    // K = k0+kh*16+8 .. +15
  return f.v;
}

// A-operand from f32 global row-major [row][k] with on-the-fly bf16 convert.
static __device__ __forceinline__ v16bf afrag_gf32(const float* S, int pitch,
                                                   int row_base, int k0, int lane) {
  const int r = lane & 15, kh = lane >> 4;
  const float* p = S + (size_t)(row_base + r) * pitch + k0 + kh * 16;
  FragU f;
#pragma unroll
  for (int e = 0; e < 8; ++e) {
    f.h[e]     = (__bf16)p[e];
    f.h[8 + e] = (__bf16)p[8 + e];
  }
  return f.v;
}

// B-operand (Kx16 tile) from bf16 LDS stored K-major per column: T[ncol][k].
// Lane holds column (lane&15); 16 contiguous K values starting at k0+kh*16.
static __device__ __forceinline__ v16bf bfrag_lds(const __bf16* T, int pitch,
                                                  int n0, int k0, int lane) {
  const int c = lane & 15, kh = lane >> 4;
  const __bf16* p = T + (n0 + c) * pitch + k0 + kh * 16;
  FragU f;
  f.q[0] = *(const v4u*)(p);
  f.q[1] = *(const v4u*)(p + 8);
  return f.v;
}

// B-operand from f32 global row-major [k][n] (n contiguous across lanes ->
// each element step is a coalesced 64B-per-half-wave read), cvt -> bf16.
static __device__ __forceinline__ v16bf bfrag_gf32(const float* S, int pitch,
                                                   int k0, int n0, int lane) {
  const int c = lane & 15, kh = lane >> 4;
  const float* p = S + (size_t)(k0 + kh * 16) * pitch + n0 + c;
  FragU f;
#pragma unroll
  for (int e = 0; e < 16; ++e) f.h[e] = (__bf16)p[(size_t)e * pitch];
  return f.v;
}

__global__ __launch_bounds__(512) void gcn_fused(
    const float* __restrict__ x,   // [8][32][256][64]
    const float* __restrict__ A0,  // [8][64][256][256]
    const float* __restrict__ A1,
    const float* __restrict__ A2,
    const float* __restrict__ W,   // [64][224]
    const float* __restrict__ bias,// [64]
    float* __restrict__ y)         // [8][64][256][64]
{
  extern __shared__ char smem[];
  __bf16* Hn = (__bf16*)smem;                          // [64][HN_PITCH]: rows 0-31 = X, 32-63 = current X1
  __bf16* Ht = (__bf16*)(smem + 64 * HN_PITCH * 2);    // [256][HT_PITCH]: H^T [node][channel]

  const int task = blockIdx.x;          // b*64 + m
  const int bb = task >> 6, mm = task & 63;
  const int tid  = threadIdx.x;
  const int lane = tid & 31;
  const int w    = tid >> 5;            // wave 0..15: owns 16-wide node tile
  const int r16  = lane & 15, hi = lane >> 4;
  const int v0   = w * 16;

  // ---- stage X_bm (strided gather over m; lines reused by 64 concurrent tasks)
  const float* xb = x + (size_t)bb * 32 * 256 * 64 + mm;
  for (int i = tid; i < 32 * 256; i += 512) {
    const int f = i >> 8, n = i & 255;
    const __bf16 hv = (__bf16)xb[(size_t)i * 64];
    Hn[f * HN_PITCH + n] = hv;
    Ht[n * HT_PITCH + f] = hv;
  }
  __syncthreads();

  for (int base = 0; base < 3; ++base) {
    const float* Ab = (base == 0 ? A0 : (base == 1 ? A1 : A2)) +
                      (size_t)task * 256 * 256;         // A[b][m][n][v]
    const int ch1 = 32 + base * 64;                     // X1 channel offset in H

    // Register-resident B-fragments of A for this wave's v-tile:
    // 8 k-steps x v16bf = 64 VGPRs, loaded from HBM exactly once,
    // reused by both the order-1 and order-2 nconv.
    v16bf bfr[8];

    // ---- nconv1: X1 = X @ A  (K = 256 over nodes)
    v8f c0 = {}, c1 = {};
#pragma unroll
    for (int k = 0; k < 8; ++k) {
      const int k0 = k * 32;
      bfr[k] = bfrag_gf32(Ab, 256, k0, v0, lane);
      const v16bf a0 = afrag_lds(Hn, HN_PITCH, 0,  k0, lane);
      const v16bf a1 = afrag_lds(Hn, HN_PITCH, 16, k0, lane);
      c0 = wmma_bf16(a0, bfr[k], c0);
      c1 = wmma_bf16(a1, bfr[k], c1);
    }
    // scatter X1 into Hn rows 32-63 (next A-operand) and Ht (for final GEMM)
#pragma unroll
    for (int g = 0; g < 8; ++g) {
      const int fr = g + 8 * hi;      // 0..15 within tile
      const int vv = v0 + r16;        // node column
      const __bf16 h0 = (__bf16)c0[g];
      const __bf16 h1 = (__bf16)c1[g];
      Hn[(32 + fr) * HN_PITCH + vv] = h0;
      Hn[(48 + fr) * HN_PITCH + vv] = h1;
      Ht[vv * HT_PITCH + ch1 + fr]      = h0;
      Ht[vv * HT_PITCH + ch1 + 16 + fr] = h1;
    }
    __syncthreads();

    // ---- nconv2: X2 = X1 @ A  (B-frags reused from registers; pure LDS+WMMA)
    v8f d0 = {}, d1 = {};
#pragma unroll
    for (int k = 0; k < 8; ++k) {
      const int k0 = k * 32;
      const v16bf a0 = afrag_lds(Hn, HN_PITCH, 32, k0, lane);
      const v16bf a1 = afrag_lds(Hn, HN_PITCH, 48, k0, lane);
      d0 = wmma_bf16(a0, bfr[k], d0);
      d1 = wmma_bf16(a1, bfr[k], d1);
    }
#pragma unroll
    for (int g = 0; g < 8; ++g) {
      const int fr = g + 8 * hi;
      const int vv = v0 + r16;
      Ht[vv * HT_PITCH + ch1 + 32 + fr] = (__bf16)d0[g];
      Ht[vv * HT_PITCH + ch1 + 48 + fr] = (__bf16)d1[g];
    }
    __syncthreads();
  }

  // ---- final 1x1 conv: Y = W[64x224] @ H[224x256] + b; wave w owns node tile w
  v8f acc0 = {}, acc1 = {}, acc2 = {}, acc3 = {};
  for (int k = 0; k < 7; ++k) {
    const int f0 = k * 32;
    const v16bf bf = bfrag_lds(Ht, HT_PITCH, v0, f0, lane);
    acc0 = wmma_bf16(afrag_gf32(W, 224, 0,  f0, lane), bf, acc0);
    acc1 = wmma_bf16(afrag_gf32(W, 224, 16, f0, lane), bf, acc1);
    acc2 = wmma_bf16(afrag_gf32(W, 224, 32, f0, lane), bf, acc2);
    acc3 = wmma_bf16(afrag_gf32(W, 224, 48, f0, lane), bf, acc3);
  }

  float* yb = y + (size_t)bb * 64 * 256 * 64 + mm;   // y[b][o][c][m]
  const int cn = v0 + r16;
#pragma unroll
  for (int g = 0; g < 8; ++g) {
    const int oo = g + 8 * hi;
    yb[((size_t)(oo)      * 256 + cn) * 64] = acc0[g] + bias[oo];
    yb[((size_t)(oo + 16) * 256 + cn) * 64] = acc1[g] + bias[oo + 16];
    yb[((size_t)(oo + 32) * 256 + cn) * 64] = acc2[g] + bias[oo + 32];
    yb[((size_t)(oo + 48) * 256 + cn) * 64] = acc3[g] + bias[oo + 48];
  }
}

extern "C" void kernel_launch(void* const* d_in, const int* in_sizes, int n_in,
                              void* d_out, int out_size, void* d_ws, size_t ws_size,
                              hipStream_t stream) {
  const float* x    = (const float*)d_in[0];
  const float* A0   = (const float*)d_in[1];
  const float* A1   = (const float*)d_in[2];
  const float* A2   = (const float*)d_in[3];
  const float* W    = (const float*)d_in[4];
  const float* bias = (const float*)d_in[5];
  float* y = (float*)d_out;

  const size_t shmem = (size_t)64 * HN_PITCH * 2 + (size_t)256 * HT_PITCH * 2; // 152576 B
  (void)hipFuncSetAttribute((const void*)gcn_fused,
                            hipFuncAttributeMaxDynamicSharedMemorySize, (int)shmem);

  gcn_fused<<<dim3(512), dim3(512), shmem, stream>>>(x, A0, A1, A2, W, bias, y);
}
